// QuantumAttention_65481071410014
// MI455X (gfx1250) — matrix operand order repaired and checked by reading, not compiled, not run on hardware
//
#include <hip/hip_runtime.h>
#include <math.h>
#include <stdint.h>

typedef __attribute__((ext_vector_type(16))) __bf16 v16bf;
typedef __attribute__((ext_vector_type(8)))  float  v8f;

#define EMBED 512
#define Tn    1024
#define Bn    2
#define ROWS  (Bn * Tn)   // 2048

union Frag { v16bf v; unsigned u[8]; };

// native f32 -> bf16 (RNE on gfx1250)
__device__ __forceinline__ __bf16 f2bf(float f) { return (__bf16)f; }

// A-operand (16x32 bf16) K element offset for dword i (0..7), hi = lane>=16
__device__ __forceinline__ int kpairA(int i, int hi) {
    return ((i & 4) << 2) + (hi << 3) + ((i & 3) << 1);   // VGPR0-3: K0-7 / +8 ; VGPR4-7: K16-23 / +8
}
// B-operand (32x16 bf16) K element offset for dword i (0..7), hi = lane>=16
__device__ __forceinline__ int kpairB(int i, int hi) {
    return ((i & 4) << 2) + (hi << 3) + ((i & 3) << 1);   // VGPR0-3: K0-7 / +8 ; VGPR4-7: K16-23 / +8
}

__device__ __forceinline__ v8f wmma_bf16(const Frag& a, const Frag& b, v8f c) {
    return __builtin_amdgcn_wmma_f32_16x16x32_bf16(false, a.v, false, b.v, (short)0, c, false, false);
}

// 16-byte global -> LDS async copy (ASYNCcnt); lptr: pointer into __shared__
__device__ __forceinline__ void async_copy16(const void* gptr, void* lptr) {
    unsigned lds = (unsigned)(uintptr_t)lptr;   // low 32 bits of generic LDS ptr = LDS byte offset
    asm volatile("global_load_async_to_lds_b128 %0, %1, off"
                 :: "v"(lds), "v"(gptr) : "memory");
}
__device__ __forceinline__ void wait_async() {
    asm volatile("s_wait_asynccnt 0" ::: "memory");
}

// ---------------------------------------------------------------------------
// Kernel 1: Q/K/V = x @ W^T  (f32 in, bf16 out; V stored transposed per batch)
// grid (M/16, N/64, 3), block 32: one wave computes a 16x64 strip (4 WMMA tiles)
// ---------------------------------------------------------------------------
__global__ __launch_bounds__(32) void qkv_kernel(
    const float* __restrict__ x, const float* __restrict__ Wq,
    const float* __restrict__ Wk, const float* __restrict__ Wv,
    __bf16* __restrict__ Qb, __bf16* __restrict__ Kb, __bf16* __restrict__ Vt) {
    const int lane = threadIdx.x;
    const int m0 = blockIdx.x * 16;            // token tile
    const int n0 = blockIdx.y * 64;            // out-dim strip
    const int which = blockIdx.z;              // 0=Q, 1=K, 2=V
    const float* W = (which == 0) ? Wq : ((which == 1) ? Wk : Wv);
    const int ml = lane & 15, hi = lane >> 4;
    const float* arow = x + (size_t)(m0 + ml) * EMBED;
    v8f acc[4] = {{}, {}, {}, {}};
    for (int kb = 0; kb < EMBED; kb += 32) {
        Frag a;
#pragma unroll
        for (int i = 0; i < 8; ++i) {
            float2 fa = *(const float2*)(arow + kb + kpairA(i, hi));
            a.v[2 * i] = f2bf(fa.x); a.v[2 * i + 1] = f2bf(fa.y);
        }
#pragma unroll
        for (int j = 0; j < 4; ++j) {
            const float* brow = W + (size_t)(n0 + 16 * j + ml) * EMBED;  // B[k][n] = W[n][k]
            Frag b;
#pragma unroll
            for (int i = 0; i < 8; ++i) {
                float2 fb = *(const float2*)(brow + kb + kpairB(i, hi));
                b.v[2 * i] = f2bf(fb.x); b.v[2 * i + 1] = f2bf(fb.y);
            }
            acc[j] = wmma_bf16(a, b, acc[j]);
        }
    }
    const int mh = hi * 8;
#pragma unroll
    for (int j = 0; j < 4; ++j) {
        const int n = n0 + 16 * j + ml;
        if (which < 2) {
            __bf16* O = (which == 0) ? Qb : Kb;
#pragma unroll
            for (int r = 0; r < 8; ++r)
                O[(size_t)(m0 + mh + r) * EMBED + n] = f2bf(acc[j][r]);
        } else {
#pragma unroll
            for (int r = 0; r < 8; ++r) {        // store V transposed: Vt[b][e][s]
                int gm = m0 + mh + r;
                int bb = gm >> 10, s = gm & (Tn - 1);
                Vt[(size_t)bb * EMBED * Tn + (size_t)n * Tn + s] = f2bf(acc[j][r]);
            }
        }
    }
}

// ---------------------------------------------------------------------------
// Kernel 2: scores[b,t,s] = (Q K^T) / sqrt(512)
// grid (T/64, T/64, B), block 128 (4 waves): 64x64 tile, LDS async staging
// ---------------------------------------------------------------------------
__global__ __launch_bounds__(128) void scores_kernel(
    const __bf16* __restrict__ Qb, const __bf16* __restrict__ Kb,
    float* __restrict__ Sc) {
    __shared__ __align__(16) unsigned char smem[8192];   // A tile 4KB | B tile 4KB
    const int tid = threadIdx.x;
    const int wv = tid >> 5, lane = tid & 31;
    const int ml = lane & 15, hi = lane >> 4, mh = hi * 8;
    const int t0 = blockIdx.x * 64, s0 = blockIdx.y * 64, b = blockIdx.z;
    const __bf16* Qbase = Qb + (size_t)b * Tn * EMBED;
    const __bf16* Kbase = Kb + (size_t)b * Tn * EMBED;
    v8f acc[4] = {{}, {}, {}, {}};
    for (int kb = 0; kb < EMBED; kb += 32) {
        if (kb) __syncthreads();                 // previous tile fully consumed
#pragma unroll
        for (int j = 0; j < 2; ++j) {            // stage A (Q rows) and B (K rows): 64 rows x 64B each
            int c = tid + 128 * j;
            int row = c >> 2, part = c & 3;
            async_copy16(Qbase + (size_t)(t0 + row) * EMBED + kb + part * 8,
                         smem + row * 64 + part * 16);
            async_copy16(Kbase + (size_t)(s0 + row) * EMBED + kb + part * 8,
                         smem + 4096 + row * 64 + part * 16);
        }
        wait_async();
        __syncthreads();
        Frag a;
#pragma unroll
        for (int i = 0; i < 8; ++i)
            a.u[i] = *(const unsigned*)(smem + (16 * wv + ml) * 64 + kpairA(i, hi) * 2);
#pragma unroll
        for (int j = 0; j < 4; ++j) {
            Frag bb;
#pragma unroll
            for (int i = 0; i < 8; ++i)
                bb.u[i] = *(const unsigned*)(smem + 4096 + (16 * j + ml) * 64 + kpairB(i, hi) * 2);
            acc[j] = wmma_bf16(a, bb, acc[j]);
        }
    }
    const float scale = 0.04419417382415922f;    // 1/sqrt(512)
    float* out = Sc + (size_t)b * Tn * Tn;
#pragma unroll
    for (int j = 0; j < 4; ++j) {
        const int s = s0 + 16 * j + ml;
#pragma unroll
        for (int r = 0; r < 8; ++r)
            out[(size_t)(t0 + 16 * wv + mh + r) * Tn + s] = acc[j][r] * scale;
    }
}

// ---------------------------------------------------------------------------
// Kernel 3: row softmax (1024 wide), write bf16 weights.  grid 2048, block 256
// ---------------------------------------------------------------------------
__global__ __launch_bounds__(256) void softmax_kernel(
    const float* __restrict__ Sc, __bf16* __restrict__ Wgt) {
    __shared__ float red[256];
    const size_t row = blockIdx.x;
    const float* r = Sc + row * Tn;
    __bf16* w = Wgt + row * Tn;
    const int tid = threadIdx.x;
    float v[4];
    float m = -INFINITY;
#pragma unroll
    for (int i = 0; i < 4; ++i) { v[i] = r[tid + 256 * i]; m = fmaxf(m, v[i]); }
    red[tid] = m; __syncthreads();
    for (int s = 128; s > 0; s >>= 1) {
        if (tid < s) red[tid] = fmaxf(red[tid], red[tid + s]);
        __syncthreads();
    }
    m = red[0]; __syncthreads();
    float sum = 0.f;
#pragma unroll
    for (int i = 0; i < 4; ++i) { v[i] = __expf(v[i] - m); sum += v[i]; }
    red[tid] = sum; __syncthreads();
    for (int s = 128; s > 0; s >>= 1) {
        if (tid < s) red[tid] += red[tid + s];
        __syncthreads();
    }
    const float inv = 1.0f / red[0];
#pragma unroll
    for (int i = 0; i < 4; ++i) w[tid + 256 * i] = f2bf(v[i] * inv);
}

// ---------------------------------------------------------------------------
// Kernel 4: 8-qubit entangling circuit on 256 amplitudes -> 16 constants
//   consts[q] = <Z_q>_psi0,  consts[8+q] = <X_q>_psi0
// ---------------------------------------------------------------------------
__global__ __launch_bounds__(256) void entangle_kernel(
    const float* __restrict__ params, float* __restrict__ consts) {
    __shared__ float psi[256];
    __shared__ float redz[256], redx[256];
    const int tid = threadIdx.x;
    psi[tid] = (tid == 0) ? 1.0f : 0.0f;
    __syncthreads();
    for (int l = 0; l < 2; ++l) {
        for (int q = 0; q < 8; ++q) {            // RY(params[l][q]) on wire q
            float th = params[l * 8 + q];
            float ch = cosf(0.5f * th), sh = sinf(0.5f * th);
            int mask = 1 << (7 - q);
            bool low = (tid & mask) == 0;
            float a0 = 0.f, a1 = 0.f;
            if (low) { a0 = psi[tid]; a1 = psi[tid | mask]; }
            __syncthreads();
            if (low) { psi[tid] = ch * a0 - sh * a1; psi[tid | mask] = sh * a0 + ch * a1; }
            __syncthreads();
        }
        for (int q = 0; q < 8; ++q) {            // CNOT(q, q+1), ... , CNOT(7, 0)
            int cm = 1 << (7 - q), tm = 1 << (7 - ((q + 1) & 7));
            bool doswap = (tid & cm) && !(tid & tm);
            float a0 = 0.f, a1 = 0.f;
            if (doswap) { a0 = psi[tid]; a1 = psi[tid | tm]; }
            __syncthreads();
            if (doswap) { psi[tid] = a1; psi[tid | tm] = a0; }
            __syncthreads();
        }
    }
    for (int q = 0; q < 8; ++q) {
        int mask = 1 << (7 - q);
        float p = psi[tid];
        redz[tid] = (tid & mask) ? -p * p : p * p;
        redx[tid] = p * psi[tid ^ mask];
        __syncthreads();
        for (int s = 128; s > 0; s >>= 1) {
            if (tid < s) { redz[tid] += redz[tid + s]; redx[tid] += redx[tid + s]; }
            __syncthreads();
        }
        if (tid == 0) { consts[q] = redz[0]; consts[8 + q] = redx[0]; }
        __syncthreads();
    }
}

// ---------------------------------------------------------------------------
// Kernel 5: out = weights @ V with fused quantum epilogue:
//   d_out[b,t,e] = cos(a)*<Z_{e%8}>_0 - sin(a)*<X_{e%8}>_0,  a = (W·V)[b,t,e]
// grid (T/64, E/64, B), block 128 (4 waves): 64x64 tile, LDS async staging
// ---------------------------------------------------------------------------
__global__ __launch_bounds__(128) void pv_kernel(
    const __bf16* __restrict__ Wgt, const __bf16* __restrict__ Vt,
    const float* __restrict__ consts, float* __restrict__ out) {
    __shared__ __align__(16) unsigned char smem[8192];   // A tile 4KB | B tile 4KB
    const int tid = threadIdx.x;
    const int wv = tid >> 5, lane = tid & 31;
    const int ml = lane & 15, hi = lane >> 4, mh = hi * 8;
    const int t0 = blockIdx.x * 64, e0 = blockIdx.y * 64, b = blockIdx.z;
    const __bf16* Abase = Wgt + (size_t)b * Tn * Tn;       // weights rows (t x s)
    const __bf16* Bbase = Vt + (size_t)b * EMBED * Tn;     // Vt rows (e x s)
    v8f acc[4] = {{}, {}, {}, {}};
    for (int kb = 0; kb < Tn; kb += 32) {
        if (kb) __syncthreads();
#pragma unroll
        for (int j = 0; j < 2; ++j) {            // stage A (weight rows) and B (Vt rows)
            int c = tid + 128 * j;
            int row = c >> 2, part = c & 3;
            async_copy16(Abase + (size_t)(t0 + row) * Tn + kb + part * 8,
                         smem + row * 64 + part * 16);
            async_copy16(Bbase + (size_t)(e0 + row) * Tn + kb + part * 8,
                         smem + 4096 + row * 64 + part * 16);
        }
        wait_async();
        __syncthreads();
        Frag a;
#pragma unroll
        for (int i = 0; i < 8; ++i)
            a.u[i] = *(const unsigned*)(smem + (16 * wv + ml) * 64 + kpairA(i, hi) * 2);
#pragma unroll
        for (int j = 0; j < 4; ++j) {
            Frag bb;
#pragma unroll
            for (int i = 0; i < 8; ++i)
                bb.u[i] = *(const unsigned*)(smem + 4096 + (16 * j + ml) * 64 + kpairB(i, hi) * 2);
            acc[j] = wmma_bf16(a, bb, acc[j]);
        }
    }
#pragma unroll
    for (int j = 0; j < 4; ++j) {
        const int e = e0 + 16 * j + ml;
        const float zc = consts[e & 7];
        const float xc = consts[8 + (e & 7)];
        float* o = out + ((size_t)b * Tn + t0 + 16 * wv + mh) * EMBED + e;
#pragma unroll
        for (int r = 0; r < 8; ++r) {
            float a = acc[j][r];
            o[(size_t)r * EMBED] = cosf(a) * zc - sinf(a) * xc;
        }
    }
}

// ---------------------------------------------------------------------------
extern "C" void kernel_launch(void* const* d_in, const int* in_sizes, int n_in,
                              void* d_out, int out_size, void* d_ws, size_t ws_size,
                              hipStream_t stream) {
    const float* x      = (const float*)d_in[0];
    const float* Wq     = (const float*)d_in[1];
    const float* Wk     = (const float*)d_in[2];
    const float* Wv     = (const float*)d_in[3];
    const float* params = (const float*)d_in[4];
    float* out = (float*)d_out;

    char* w = (char*)d_ws;
    size_t off = 0;
    __bf16* Qb = (__bf16*)(w + off); off += (size_t)ROWS * EMBED * 2;       // 2 MB
    __bf16* Kb = (__bf16*)(w + off); off += (size_t)ROWS * EMBED * 2;       // 2 MB
    __bf16* Vt = (__bf16*)(w + off); off += (size_t)Bn * EMBED * Tn * 2;    // 2 MB
    float*  Sc = (float*)(w + off);  off += (size_t)Bn * Tn * Tn * 4;       // 8 MB
    __bf16* Wgt = (__bf16*)(w + off); off += (size_t)Bn * Tn * Tn * 2;      // 4 MB
    float*  consts = (float*)(w + off);                                     // 64 B

    entangle_kernel<<<1, 256, 0, stream>>>(params, consts);
    qkv_kernel<<<dim3(ROWS / 16, EMBED / 64, 3), 32, 0, stream>>>(x, Wq, Wk, Wv, Qb, Kb, Vt);
    scores_kernel<<<dim3(Tn / 64, Tn / 64, Bn), 128, 0, stream>>>(Qb, Kb, Sc);
    softmax_kernel<<<ROWS, 256, 0, stream>>>(Sc, Wgt);
    pv_kernel<<<dim3(Tn / 64, EMBED / 64, Bn), 128, 0, stream>>>(Wgt, Vt, consts, out);
}
